// RetrieverBase_27092653703271
// MI455X (gfx1250) — matrix operand order repaired and checked by reading, not compiled, not run on hardware
//
#include <hip/hip_runtime.h>

// ---------------------------------------------------------------------------
// ColBERT-style in-batch late-interaction scoring for MI455X (gfx1250).
//   term_relevance[a,b,q,d] = sum_v q[a,q,v] * p[b,d,v]     (A=64,B=128,Q=32,D=180,V=128)
//   query_maxsim[a,b,q]     = max_d term_relevance
//   relevance[a,b]          = sum_q query_maxsim
//
// v2: split-f16 3-product GEMM on v_wmma_f32_16x16x32_f16.
//   x = hi + lo (hi = f16(x), lo = f16(x-hi));  A*B ~= Ah*Bh + Ah*Bl + Al*Bh
//   (dropped Al*Bl term ~2^-22 relative -> fp32-quality vs f32 reference).
// Prologue kernel converts q,p once into hi/lo f16 arrays in d_ws (12.25 MB,
// L2-resident). Main kernel: one block per (a,b), 6 waves x (32x32 tile),
// 12 f16 WMMAs per K=32 chunk (48 per wave) vs 128 f32 WMMAs in v1.
// 190 MB term_relevance streamed with non-temporal stores (store floor ~8.2us
// @ 23.3 TB/s is the target bound; f16 matrix compute now sits at/below it).
// ---------------------------------------------------------------------------

typedef _Float16 v8h  __attribute__((ext_vector_type(8)));
typedef _Float16 v16h __attribute__((ext_vector_type(16)));
typedef float    v8f  __attribute__((ext_vector_type(8)));

#define A_N 64
#define B_N 128
#define Q_N 32
#define D_N 180
#define V_N 128
#define NWAVE 6
#define NTHR  (NWAVE * 32)
#define QSH 136                    // LDS row stride in halves: 272B -> bank-conflict-free b128
#define NQ_ELEM (A_N * Q_N * V_N)  // 262144
#define NP_ELEM (B_N * D_N * V_N)  // 2949120

// ---- prologue: one-time fp32 -> (hi,lo) f16 split of q and p ----
__global__ __launch_bounds__(256)
void split_f16_kernel(const float* __restrict__ q, const float* __restrict__ p,
                      _Float16* __restrict__ qh, _Float16* __restrict__ ql,
                      _Float16* __restrict__ ph, _Float16* __restrict__ pl)
{
    int i = blockIdx.x * blockDim.x + threadIdx.x;
    if (i < NQ_ELEM) {
        float x = q[i];
        _Float16 h = (_Float16)x;
        qh[i] = h;
        ql[i] = (_Float16)(x - (float)h);
    }
    if (i < NP_ELEM) {
        float x = p[i];
        _Float16 h = (_Float16)x;
        ph[i] = h;
        pl[i] = (_Float16)(x - (float)h);
    }
}

// A fragment (16x32 f16): two contiguous 8-half groups (K=k0+8*hi+j , K=k0+16+8*hi+j)
__device__ __forceinline__ v16h ld_frag2(const _Float16* p0, const _Float16* p1)
{
    v8h x = *(const v8h*)p0;
    v8h y = *(const v8h*)p1;
    return __builtin_shufflevector(x, y, 0, 1, 2, 3, 4, 5, 6, 7,
                                         8, 9, 10, 11, 12, 13, 14, 15);
}

__global__ __launch_bounds__(NTHR)
void colbert_score_kernel(const _Float16* __restrict__ qh, const _Float16* __restrict__ ql,
                          const _Float16* __restrict__ ph, const _Float16* __restrict__ pl,
                          float* __restrict__ out_tr,
                          float* __restrict__ out_max,
                          float* __restrict__ out_rel)
{
    __shared__ __align__(16) _Float16 qsh[Q_N * QSH];  // q hi, padded rows (~8.7 KB)
    __shared__ __align__(16) _Float16 qsl[Q_N * QSH];  // q lo
    __shared__ float wave_max[NWAVE][Q_N];

    const int ab = blockIdx.x;           // 0 .. A*B-1
    const int a  = ab >> 7;              // / B_N
    const int b  = ab & (B_N - 1);

    const int tid  = threadIdx.x;
    const int lane = tid & 31;
    const int w    = tid >> 5;           // wave 0..5
    const int li   = lane & 15;          // lane within half-group
    const int hi   = lane >> 4;          // K-half (A/B frags), M-half (C frag)

    // ---- stage q[a] hi/lo f16 (16 KB) into LDS (16B vector copies) ----
    {
        const uint4* gh = (const uint4*)(qh + (size_t)a * (Q_N * V_N));
        const uint4* gl = (const uint4*)(ql + (size_t)a * (Q_N * V_N));
        for (int i = tid; i < (Q_N * V_N / 8); i += NTHR) {   // 512 x uint4 per array
            int row  = i >> 4;            // 16 uint4 (128 halves) per row
            int col8 = (i & 15) << 3;     // half offset
            *(uint4*)&qsh[row * QSH + col8] = gh[i];
            *(uint4*)&qsl[row * QSH + col8] = gl[i];
        }
    }
    __syncthreads();

    // ---- per-wave 32x32 tile ----
    const int n_base = w * 32;
    const int c0  = n_base + li;                    // always < 180 (max 175)
    const int c1  = n_base + 16 + li;               // may exceed 179 for wave 5
    const int c1c = (c1 < D_N) ? c1 : (D_N - 1);    // clamp address: EXEC stays uniform

    // A-frag LDS bases: row = mi*16 + li, first group at K = 8*hi (+k0), second at +16
    const _Float16* aH0 = qsh + (0  + li) * QSH + (hi << 3);
    const _Float16* aL0 = qsl + (0  + li) * QSH + (hi << 3);
    const _Float16* aH1 = qsh + (16 + li) * QSH + (hi << 3);
    const _Float16* aL1 = qsl + (16 + li) * QSH + (hi << 3);
    // B-frag global bases: row = doc token, first group at K = 8*hi (+k0), second at +16
    const size_t bo0 = (size_t)(b * D_N + c0 ) * V_N + (hi << 3);
    const size_t bo1 = (size_t)(b * D_N + c1c) * V_N + (hi << 3);
    const _Float16* bH0 = ph + bo0;
    const _Float16* bL0 = pl + bo0;
    const _Float16* bH1 = ph + bo1;
    const _Float16* bL1 = pl + bo1;

    v8f acc00 = {}, acc01 = {}, acc10 = {}, acc11 = {};

    #pragma unroll
    for (int k0 = 0; k0 < V_N; k0 += 32) {
        v16h a0h = ld_frag2(aH0 + k0, aH0 + k0 + 16);
        v16h a0l = ld_frag2(aL0 + k0, aL0 + k0 + 16);
        v16h a1h = ld_frag2(aH1 + k0, aH1 + k0 + 16);
        v16h a1l = ld_frag2(aL1 + k0, aL1 + k0 + 16);
        v16h b0h = ld_frag2(bH0 + k0, bH0 + k0 + 16);
        v16h b0l = ld_frag2(bL0 + k0, bL0 + k0 + 16);
        v16h b1h = ld_frag2(bH1 + k0, bH1 + k0 + 16);
        v16h b1l = ld_frag2(bL1 + k0, bL1 + k0 + 16);

        acc00 = __builtin_amdgcn_wmma_f32_16x16x32_f16(false, a0h, false, b0h, (short)0, acc00, false, false);
        acc00 = __builtin_amdgcn_wmma_f32_16x16x32_f16(false, a0h, false, b0l, (short)0, acc00, false, false);
        acc00 = __builtin_amdgcn_wmma_f32_16x16x32_f16(false, a0l, false, b0h, (short)0, acc00, false, false);

        acc01 = __builtin_amdgcn_wmma_f32_16x16x32_f16(false, a0h, false, b1h, (short)0, acc01, false, false);
        acc01 = __builtin_amdgcn_wmma_f32_16x16x32_f16(false, a0h, false, b1l, (short)0, acc01, false, false);
        acc01 = __builtin_amdgcn_wmma_f32_16x16x32_f16(false, a0l, false, b1h, (short)0, acc01, false, false);

        acc10 = __builtin_amdgcn_wmma_f32_16x16x32_f16(false, a1h, false, b0h, (short)0, acc10, false, false);
        acc10 = __builtin_amdgcn_wmma_f32_16x16x32_f16(false, a1h, false, b0l, (short)0, acc10, false, false);
        acc10 = __builtin_amdgcn_wmma_f32_16x16x32_f16(false, a1l, false, b0h, (short)0, acc10, false, false);

        acc11 = __builtin_amdgcn_wmma_f32_16x16x32_f16(false, a1h, false, b1h, (short)0, acc11, false, false);
        acc11 = __builtin_amdgcn_wmma_f32_16x16x32_f16(false, a1h, false, b1l, (short)0, acc11, false, false);
        acc11 = __builtin_amdgcn_wmma_f32_16x16x32_f16(false, a1l, false, b1h, (short)0, acc11, false, false);
    }

    // ---- stream term_relevance (non-temporal) ----
    // C/D layout: VGPR r -> M = r + 8*hi (+16*mi), N = li
    const size_t tr_base = (size_t)ab * (Q_N * D_N);
    const bool c1ok = (c1 < D_N);
    #pragma unroll
    for (int r = 0; r < 8; ++r) {
        const int row = r + (hi << 3);
        __builtin_nontemporal_store(acc00[r], &out_tr[tr_base + (size_t)(row     ) * D_N + c0]);
        __builtin_nontemporal_store(acc10[r], &out_tr[tr_base + (size_t)(row + 16) * D_N + c0]);
        if (c1ok) {
            __builtin_nontemporal_store(acc01[r], &out_tr[tr_base + (size_t)(row     ) * D_N + c1]);
            __builtin_nontemporal_store(acc11[r], &out_tr[tr_base + (size_t)(row + 16) * D_N + c1]);
        }
    }

    // ---- per-wave max over its 32 doc columns, per query row ----
    const float NEG = -3.402823466e38f;
    #pragma unroll
    for (int mi = 0; mi < 2; ++mi) {
        #pragma unroll
        for (int r = 0; r < 8; ++r) {
            float v0 = (mi == 0) ? acc00[r] : acc10[r];
            float v1 = (mi == 0) ? acc01[r] : acc11[r];
            float x = fmaxf(v0, c1ok ? v1 : NEG);
            x = fmaxf(x, __shfl_xor(x, 8, 32));   // 16-lane half-group reduction
            x = fmaxf(x, __shfl_xor(x, 4, 32));
            x = fmaxf(x, __shfl_xor(x, 2, 32));
            x = fmaxf(x, __shfl_xor(x, 1, 32));
            if (li == 0) wave_max[w][mi * 16 + r + (hi << 3)] = x;
        }
    }
    __syncthreads();

    // ---- combine 6 waves -> query_maxsim; 32-lane sum -> relevance ----
    if (tid < 32) {
        float mx = wave_max[0][tid];
        #pragma unroll
        for (int ww = 1; ww < NWAVE; ++ww) mx = fmaxf(mx, wave_max[ww][tid]);
        out_max[(size_t)ab * Q_N + tid] = mx;
        float s = mx;
        s += __shfl_xor(s, 16, 32);
        s += __shfl_xor(s, 8, 32);
        s += __shfl_xor(s, 4, 32);
        s += __shfl_xor(s, 2, 32);
        s += __shfl_xor(s, 1, 32);
        if (tid == 0) out_rel[ab] = s;
    }
}

extern "C" void kernel_launch(void* const* d_in, const int* in_sizes, int n_in,
                              void* d_out, int out_size, void* d_ws, size_t ws_size,
                              hipStream_t stream) {
    const float* q = (const float*)d_in[0];   // [A, Q, V] f32
    const float* p = (const float*)d_in[1];   // [B, D, V] f32

    // workspace layout: qh | ql | ph | pl  (f16), total 12.25 MB
    _Float16* qh = (_Float16*)d_ws;
    _Float16* ql = qh + NQ_ELEM;
    _Float16* ph = ql + NQ_ELEM;
    _Float16* pl = ph + NP_ELEM;

    float* out     = (float*)d_out;
    float* out_tr  = out;                                         // [A,B,Q,D]
    float* out_max = out_tr  + (size_t)A_N * B_N * Q_N * D_N;     // [A,B,Q]
    float* out_rel = out_max + (size_t)A_N * B_N * Q_N;           // [A,B]

    split_f16_kernel<<<dim3((NP_ELEM + 255) / 256), dim3(256), 0, stream>>>(
        q, p, qh, ql, ph, pl);

    colbert_score_kernel<<<dim3(A_N * B_N), dim3(NTHR), 0, stream>>>(
        qh, ql, ph, pl, out_tr, out_max, out_rel);
}
